// PGExplainerStyle_9483287790247
// MI455X (gfx1250) — matrix operand order repaired and checked by reading, not compiled, not run on hardware
//
#include <hip/hip_runtime.h>
#include <hip/hip_bf16.h>

// CDNA5 / gfx1250: wave32, WMMA bf16 path, LDS-staged B, software-pipelined.
typedef __attribute__((ext_vector_type(16))) __bf16       v16bf;
typedef __attribute__((ext_vector_type(8)))  float        v8f;
typedef __attribute__((ext_vector_type(4)))  unsigned int v4u;

#define HIDDEN   128
#define ATTR_DIM 16
#define IN_DIM   (2 * HIDDEN + ATTR_DIM)   // 272
#define KSTEPS   9                          // ceil(272/32) -> K padded to 288
#define NTILES   (HIDDEN / 16)              // 8
#define FRAGS_TOTAL (KSTEPS * NTILES * 32)  // 2304 lane-fragments of 16 bf16
#define CHUNKS_TOTAL (FRAGS_TOTAL * 2)      // 4608 x 16-byte chunks (73,728 B)

// ---------------------------------------------------------------------------
// Prep: swizzle W1 [272,128] f32 into per-lane bf16 WMMA B-fragments.
// B layout (16-bit B 32x16): lane L -> column N = L&15; halves h=0..15 hold
// rows K = (L>>4)*8 + (h<8 ? h : h+8) of the 32-row K-slab.
// ---------------------------------------------------------------------------
__global__ void pg_prep_w1_frags(const float* __restrict__ W1,
                                 v16bf* __restrict__ frags) {
  int tid = blockIdx.x * blockDim.x + threadIdx.x;
  if (tid >= FRAGS_TOTAL) return;
  const int lane = tid & 31;
  const int nt   = (tid >> 5) & (NTILES - 1);
  const int t    = tid >> 8;
  const int N    = (nt << 4) + (lane & 15);
  const int kbase = t * 32 + ((lane >> 4) << 3);
  v16bf v;
#pragma unroll
  for (int h = 0; h < 16; ++h) {
    const int K = kbase + (h < 8 ? h : h + 8);
    const float f = (K < IN_DIM) ? W1[(size_t)K * HIDDEN + N] : 0.0f;
    v[h] = (__bf16)f;
  }
  frags[tid] = v;
}

// ---------------------------------------------------------------------------
// Raw (f32) A-tile data for one K-step: two 8-float runs per lane.
// Halves p*8+q of the fragment hold K = t*32 + p*16 + khalf*8 + q; all
// segment boundaries (128/256/272) are multiples of 16, so with t unrolled
// the segment choice resolves at compile time.
// ---------------------------------------------------------------------------
struct RawA { float4 r0, r1, r2, r3; };

__device__ __forceinline__ RawA pg_load_rawA(const float* __restrict__ srcRow,
                                             const float* __restrict__ dstRow,
                                             const float* __restrict__ attRow,
                                             int kh8, int t) {
  RawA r;
#pragma unroll
  for (int p = 0; p < 2; ++p) {
    const int X0 = t * 32 + p * 16;          // compile-time after unroll
    float4 lo, hi;
    if (X0 < HIDDEN) {
      const float* q = srcRow + X0 + kh8;
      lo = *(const float4*)q;
      hi = *(const float4*)(q + 4);
    } else if (X0 < 2 * HIDDEN) {
      const float* q = dstRow + (X0 - HIDDEN) + kh8;
      lo = *(const float4*)q;
      hi = *(const float4*)(q + 4);
    } else if (X0 < IN_DIM) {
      const float* q = attRow + (X0 - 2 * HIDDEN) + kh8;
      lo = *(const float4*)q;
      hi = *(const float4*)(q + 4);
    } else {
      lo = make_float4(0.f, 0.f, 0.f, 0.f);
      hi = lo;
    }
    if (p == 0) { r.r0 = lo; r.r1 = hi; } else { r.r2 = lo; r.r3 = hi; }
  }
  return r;
}

__device__ __forceinline__ v16bf pg_cvt_afrag(const RawA& r) {
  v16bf a;
  a[0]  = (__bf16)r.r0.x;  a[1]  = (__bf16)r.r0.y;
  a[2]  = (__bf16)r.r0.z;  a[3]  = (__bf16)r.r0.w;
  a[4]  = (__bf16)r.r1.x;  a[5]  = (__bf16)r.r1.y;
  a[6]  = (__bf16)r.r1.z;  a[7]  = (__bf16)r.r1.w;
  a[8]  = (__bf16)r.r2.x;  a[9]  = (__bf16)r.r2.y;
  a[10] = (__bf16)r.r2.z;  a[11] = (__bf16)r.r2.w;
  a[12] = (__bf16)r.r3.x;  a[13] = (__bf16)r.r3.y;
  a[14] = (__bf16)r.r3.z;  a[15] = (__bf16)r.r3.w;
  return a;
}

// ---------------------------------------------------------------------------
// Main: one wave = 32 edges (two 16-row M-tiles), 144 WMMAs/wave.
// Pipelining: A raw loads for step t+1 issue before the WMMA block of step t;
// B fragments double-buffer so each ds wait is covered by two WMMAs.
// ---------------------------------------------------------------------------
__global__ __launch_bounds__(256) void pg_edge_mlp_wmma(
    const float* __restrict__ node_emb,       // [N_NODES,128]
    const long long* __restrict__ edge_index, // [2,E] int64
    const float* __restrict__ edge_attr,      // [E,16]
    const float* __restrict__ b1,             // [128]
    const float* __restrict__ W2,             // [128]
    const float* __restrict__ b2,             // [1]
    const v4u* __restrict__ fragChunks,       // pre-swizzled W1 (bf16), 16B chunks
    float* __restrict__ out,                  // [E]
    int E) {
  // 73,728 B of LDS: all of W1 in WMMA-B fragment order, 16B-chunk swizzled.
  __shared__ v4u ldsB[CHUNKS_TOTAL];

  const int tid = threadIdx.x;
  // Cooperative fill with XOR swizzle: chunk w within a 64-chunk (t,nt) group
  // is stored at w ^ ((w>>4)&1) so ds_load_b128 reads are bank-conflict free.
#pragma unroll
  for (int i = tid; i < CHUNKS_TOTAL; i += 256) {
    const int g = i >> 6;
    const int w = i & 63;
    ldsB[(g << 6) + (w ^ ((w >> 4) & 1))] = fragChunks[i];
  }
  __syncthreads();

  const int lane  = tid & 31;
  const int wave  = tid >> 5;
  const int ebase = (blockIdx.x * 8 + wave) * 32;   // 32 edges per wave
  const int row   = lane & 15;
  const int khalf = lane >> 4;
  const int kh8   = khalf * 8;
  // Swizzled chunk indices for this lane's B fragment halves.
  const int sw = (lane >> 3) & 1;
  const int c0 = lane * 2 + sw;        // halves 0..7
  const int c1 = lane * 2 + 1 - sw;    // halves 8..15

  // Two M-tiles of 16 edges each.
  const int e0 = ebase + row;
  const int e1 = ebase + 16 + row;
  const int ec0 = (e0 < E) ? e0 : (E - 1);
  const int ec1 = (e1 < E) ? e1 : (E - 1);
  const int src0 = (int)edge_index[ec0];
  const int dst0 = (int)edge_index[(size_t)E + ec0];
  const int src1 = (int)edge_index[ec1];
  const int dst1 = (int)edge_index[(size_t)E + ec1];
  const float* __restrict__ srcRow0 = node_emb + (size_t)src0 * HIDDEN;
  const float* __restrict__ dstRow0 = node_emb + (size_t)dst0 * HIDDEN;
  const float* __restrict__ attRow0 = edge_attr + (size_t)ec0 * ATTR_DIM;
  const float* __restrict__ srcRow1 = node_emb + (size_t)src1 * HIDDEN;
  const float* __restrict__ dstRow1 = node_emb + (size_t)dst1 * HIDDEN;
  const float* __restrict__ attRow1 = edge_attr + (size_t)ec1 * ATTR_DIM;

  v8f acc0[NTILES];
  v8f acc1[NTILES];
#pragma unroll
  for (int i = 0; i < NTILES; ++i) {
    acc0[i] = (v8f){0.f, 0.f, 0.f, 0.f, 0.f, 0.f, 0.f, 0.f};
    acc1[i] = (v8f){0.f, 0.f, 0.f, 0.f, 0.f, 0.f, 0.f, 0.f};
  }

  // Prologue: issue A raw loads for step 0.
  RawA raw0 = pg_load_rawA(srcRow0, dstRow0, attRow0, kh8, 0);
  RawA raw1 = pg_load_rawA(srcRow1, dstRow1, attRow1, kh8, 0);

  union BU { v4u q[2]; v16bf v; };

#pragma unroll
  for (int t = 0; t < KSTEPS; ++t) {
    // Convert this step's A data (loads were issued an iteration ago).
    const v16bf a0 = pg_cvt_afrag(raw0);
    const v16bf a1 = pg_cvt_afrag(raw1);
    // Prefetch next step's A raw data before the WMMA block.
    if (t + 1 < KSTEPS) {
      raw0 = pg_load_rawA(srcRow0, dstRow0, attRow0, kh8, t + 1);
      raw1 = pg_load_rawA(srcRow1, dstRow1, attRow1, kh8, t + 1);
    }

    // B: double-buffered LDS reads.
    BU bcur, bnxt;
    {
      const int base = (t * NTILES) << 6;
      bcur.q[0] = ldsB[base + c0];
      bcur.q[1] = ldsB[base + c1];
    }
#pragma unroll
    for (int nt = 0; nt < NTILES; ++nt) {
      if (nt + 1 < NTILES) {
        const int base = (t * NTILES + nt + 1) << 6;
        bnxt.q[0] = ldsB[base + c0];
        bnxt.q[1] = ldsB[base + c1];
      }
      acc0[nt] = __builtin_amdgcn_wmma_f32_16x16x32_bf16(
          false, a0, false, bcur.v, (short)0, acc0[nt], false, false);
      acc1[nt] = __builtin_amdgcn_wmma_f32_16x16x32_bf16(
          false, a1, false, bcur.v, (short)0, acc1[nt], false, false);
      if (nt + 1 < NTILES) bcur = bnxt;
    }
  }

  // Layer 2: out[e] = sum_n relu(h[e,n] + b1[n]) * W2[n] + b2.
  // D layout: acc[nt][r] is (M = khalf*8 + r, N = nt*16 + (lane&15)).
  float partial0[8], partial1[8];
#pragma unroll
  for (int r = 0; r < 8; ++r) { partial0[r] = 0.f; partial1[r] = 0.f; }
  const int ncol = lane & 15;
#pragma unroll
  for (int nt = 0; nt < NTILES; ++nt) {
    const int N = nt * 16 + ncol;
    const float bb = b1[N];
    const float w2 = W2[N];
#pragma unroll
    for (int r = 0; r < 8; ++r) {
      float h0 = acc0[nt][r] + bb;
      float h1 = acc1[nt][r] + bb;
      h0 = (h0 > 0.f) ? h0 : 0.f;
      h1 = (h1 > 0.f) ? h1 : 0.f;
      partial0[r] = fmaf(h0, w2, partial0[r]);
      partial1[r] = fmaf(h1, w2, partial1[r]);
    }
  }
  // Reduce across the 16 lanes of this half-wave (masks stay in-group).
#pragma unroll
  for (int m = 8; m >= 1; m >>= 1) {
#pragma unroll
    for (int r = 0; r < 8; ++r) {
      partial0[r] += __shfl_xor(partial0[r], m, 32);
      partial1[r] += __shfl_xor(partial1[r], m, 32);
    }
  }
  if (ncol == 0) {
    const float bias2 = b2[0];
#pragma unroll
    for (int r = 0; r < 8; ++r) {
      const int eo0 = ebase + khalf * 8 + r;
      const int eo1 = ebase + 16 + khalf * 8 + r;
      if (eo0 < E) out[eo0] = partial0[r] + bias2;
      if (eo1 < E) out[eo1] = partial1[r] + bias2;
    }
  }
}

// ---------------------------------------------------------------------------
extern "C" void kernel_launch(void* const* d_in, const int* in_sizes, int n_in,
                              void* d_out, int out_size, void* d_ws, size_t ws_size,
                              hipStream_t stream) {
  const float*     node_emb   = (const float*)d_in[0];
  const long long* edge_index = (const long long*)d_in[1];   // int64 [2,E]
  const float*     edge_attr  = (const float*)d_in[2];
  const float*     W1         = (const float*)d_in[3];       // [272,128]
  const float*     b1         = (const float*)d_in[4];
  const float*     W2         = (const float*)d_in[5];
  const float*     b2         = (const float*)d_in[6];
  float*           out        = (float*)d_out;

  const int E = in_sizes[1] / 2;
  v16bf* frags = (v16bf*)d_ws;   // 2304 * 32 B = 73,728 B of scratch

  pg_prep_w1_frags<<<(FRAGS_TOTAL + 255) / 256, 256, 0, stream>>>(W1, frags);

  const int blocks = (E + 255) / 256;        // 8 waves x 32 edges per block
  pg_edge_mlp_wmma<<<blocks, 256, 0, stream>>>(
      node_emb, edge_index, edge_attr, b1, W2, b2, (const v4u*)frags, out, E);
}
